// SchNet_46943992546011
// MI455X (gfx1250) — matrix operand order repaired and checked by reading, not compiled, not run on hardware
//
#include <hip/hip_runtime.h>
#include <hip/hip_bf16.h>
#include <math.h>

// ---------------------------------------------------------------------------
// SchNet forward for MI455X (gfx1250, wave32, WMMA f16->f32)
// ---------------------------------------------------------------------------

typedef __attribute__((ext_vector_type(16))) _Float16 v16h;
typedef __attribute__((ext_vector_type(8)))  _Float16 v8h;
typedef __attribute__((ext_vector_type(4)))  _Float16 v4h;
typedef __attribute__((ext_vector_type(2)))  _Float16 v2h;
typedef __attribute__((ext_vector_type(8)))  float    v8f;

#define H_     128
#define F_     128
#define G_     50
#define LDSTR  136    // LDS row stride in halves (272B: 16B-aligned rows, bank spread)
#define WARPS  8      // waves per 256-thread block

#define LOG2E_F  1.44269504089f
#define LN2_F    0.69314718056f

// Branch-free shifted softplus on raw HW transcendentals.
// softplus(x) = max(x,0) + log(1+exp(-|x|));  argument of log is in [1,2],
// never denormal, so bare v_log_f32 (base-2) is safe:
//   ssp(x) = max(x,0) + ln2 * (log2(1 + 2^(-|x|*log2e)) - 1)
__device__ __forceinline__ float sspf(float x) {
    float t = __builtin_amdgcn_exp2f(-fabsf(x) * LOG2E_F);
    float l = __builtin_amdgcn_logf(1.0f + t);           // log2
    return fmaf(LN2_F, l - 1.0f, fmaxf(x, 0.0f));
}

// A-fragment (16x32 f16, M x K): lane m=lane&15, kh=lane>>4
//   a[i]   = A[m][kh*8 + i]   a[i+8] = A[m][16 + kh*8 + i]
// Loaded as two aligned 16B chunks (ds_load_b128).
__device__ __forceinline__ v16h load_a_frag(const _Float16* p, int stride, int lane) {
    int m = lane & 15, kh = lane >> 4;
    v8h lo = *(const v8h*)(p + m * stride + kh * 8);
    v8h hi = *(const v8h*)(p + m * stride + 16 + kh * 8);
    v16h a;
#pragma unroll
    for (int i = 0; i < 8; ++i) { a[i] = lo[i]; a[i + 8] = hi[i]; }
    return a;
}

__device__ __forceinline__ v8f wmma16(v16h a, v16h b, v8f c) {
    return __builtin_amdgcn_wmma_f32_16x16x32_f16(
        /*neg_a=*/false, a, /*neg_b=*/false, b,
        /*c_mod=*/(short)0, c, /*reuse_a=*/false, /*reuse_b=*/false);
}

__device__ __forceinline__ void waitDS() {
    asm volatile("s_wait_dscnt 0x0" ::: "memory");
}

// Preload a [Ksrc x 128] f32 weight into LDS in *fragment-major* f16 layout:
// frag[((kt*8 + nt)*32 + lane)*16 + j]  =  B[kt*32 + (j>>3)*16 + (lane>>4)*8 + (j&7)][nt*16 + (lane&15)]
// so each lane's 16 B-fragment halves are one contiguous 32-byte LDS chunk.
__device__ __forceinline__ void preload_b_frag(_Float16* frag, const float* w,
                                               int nkt, int Ksrc,
                                               int tid, int nthreads) {
    int total = nkt * 8 * 32 * 16;
    for (int i = tid; i < total; i += nthreads) {
        int j  = i & 15;
        int ln = (i >> 4) & 31;
        int nt = (i >> 9) & 7;
        int kt = i >> 12;
        int k  = kt * 32 + (j >> 3) * 16 + (ln >> 4) * 8 + (j & 7);
        int n  = nt * 16 + (ln & 15);
        frag[i] = (k < Ksrc) ? (_Float16)w[k * F_ + n] : (_Float16)0.0f;
    }
}

__device__ __forceinline__ v16h load_b_frag(const _Float16* frag, int kt, int nt, int lane) {
    return *(const v16h*)(frag + (((kt << 3) | nt) * 32 + lane) * 16);
}

// ---------------------------------------------------------------------------
// h = emb[atom_charges]      [N,128] f32
// ---------------------------------------------------------------------------
__global__ void gather_emb_kernel(const int* __restrict__ q,
                                  const float* __restrict__ emb,
                                  float* __restrict__ h, int n) {
    int idx = blockIdx.x * blockDim.x + threadIdx.x;   // n*32 float4 chunks
    if (idx < n * 32) {
        int node = idx >> 5, c = idx & 31;
        ((float4*)h)[(size_t)node * 32 + c] =
            ((const float4*)emb)[(size_t)q[node] * 32 + c];
    }
}

__global__ void zero_f32_kernel(float* __restrict__ p, int n) {
    int i = blockIdx.x * blockDim.x + threadIdx.x;
    if (i < n) p[i] = 0.0f;
}

// ---------------------------------------------------------------------------
// Edge filter: W[e] = ssp(gauss(d_e) @ w_e1 + b_e1) @ w_e2 + b_e2   -> f16
// One wave per 16-edge tile.
// ---------------------------------------------------------------------------
__global__ void __launch_bounds__(256)
filter_kernel(const float* __restrict__ coords,
              const int* __restrict__ ei,          // [2,E]
              const float* __restrict__ w_e1,      // [50,128]
              const float* __restrict__ b_e1,      // [128]
              const float* __restrict__ w_e2,      // [128,128]
              const float* __restrict__ b_e2,      // [128]
              _Float16* __restrict__ Wout,         // [E,128] f16
              int E_) {
    extern __shared__ __align__(32) char smem_raw[];
    _Float16* smem = (_Float16*)smem_raw;
    _Float16* w1f = smem;                        // 2*8*32*16 = 8192 halves
    _Float16* w2f = w1f + 2 * 8 * 32 * 16;       // 4*8*32*16 = 16384 halves
    const int tid  = threadIdx.x;
    const int lane = tid & 31;
    const int warp = tid >> 5;
    _Float16* buf0 = w2f + 4 * 8 * 32 * 16 + warp * (2 * 16 * LDSTR);
    _Float16* buf1 = buf0 + 16 * LDSTR;

    preload_b_frag(w1f, w_e1, 2, G_, tid, blockDim.x);
    preload_b_frag(w2f, w_e2, 4, F_, tid, blockDim.x);
    __syncthreads();

    const float delta  = 10.0f / 49.0f;              // linspace(0,10,50) step
    const float coeff2 = (-0.5f / (delta * delta)) * LOG2E_F;  // exp -> exp2

    int wave   = (blockIdx.x * blockDim.x + tid) >> 5;
    int nwave  = (gridDim.x * blockDim.x) >> 5;
    int ntiles = (E_ + 15) >> 4;
    const int ncol = lane & 15, mh = lane >> 4;

    for (int t = wave; t < ntiles; t += nwave) {
        int e0 = t << 4;

        // ---- distances: lanes 0..15 each compute one edge, broadcast via shfl
        float d;
        {
            int e = e0 + (lane & 15);
            if (e >= E_) e = E_ - 1;
            int r = ei[e], c = ei[E_ + e];
            float dx = coords[3 * r]     - coords[3 * c];
            float dy = coords[3 * r + 1] - coords[3 * c + 1];
            float dz = coords[3 * r + 2] - coords[3 * c + 2];
            d = sqrtf(dx * dx + dy * dy + dz * dz);
        }
        d = __shfl(d, lane & 15, 32);

        // ---- gaussian smearing -> buf0[16][64] (zero-padded 50..63), packed v2h
        {
            int m = lane & 15, kh = lane >> 4;
#pragma unroll
            for (int g = 0; g < 16; ++g) {
                int gg = kh * 32 + 2 * g;
                float u0 = d - gg * delta;
                float u1 = d - (gg + 1) * delta;
                float p0 = (gg     < G_) ? __builtin_amdgcn_exp2f(coeff2 * u0 * u0) : 0.0f;
                float p1 = (gg + 1 < G_) ? __builtin_amdgcn_exp2f(coeff2 * u1 * u1) : 0.0f;
                v2h pv = {(_Float16)p0, (_Float16)p1};
                *(v2h*)(buf0 + m * LDSTR + gg) = pv;
            }
        }
        waitDS();

        // ---- layer 1: [16x64] @ [64x128], ssp epilogue -> buf1 (f16)
#pragma unroll
        for (int nt = 0; nt < 8; ++nt) {
            v8f acc = {0, 0, 0, 0, 0, 0, 0, 0};
#pragma unroll
            for (int kt = 0; kt < 2; ++kt) {
                v16h a = load_a_frag(buf0 + kt * 32, LDSTR, lane);
                v16h b = load_b_frag(w1f, kt, nt, lane);
                acc = wmma16(a, b, acc);
            }
            float bb = b_e1[nt * 16 + ncol];
#pragma unroll
            for (int r = 0; r < 8; ++r)
                buf1[(r + 8 * mh) * LDSTR + nt * 16 + ncol] =
                    (_Float16)sspf(acc[r] + bb);
        }
        waitDS();

        // ---- layer 2: [16x128] @ [128x128] -> buf0 (f16)
#pragma unroll
        for (int nt = 0; nt < 8; ++nt) {
            v8f acc = {0, 0, 0, 0, 0, 0, 0, 0};
#pragma unroll
            for (int kt = 0; kt < 4; ++kt) {
                v16h a = load_a_frag(buf1 + kt * 32, LDSTR, lane);
                v16h b = load_b_frag(w2f, kt, nt, lane);
                acc = wmma16(a, b, acc);
            }
            float bb = b_e2[nt * 16 + ncol];
#pragma unroll
            for (int r = 0; r < 8; ++r)
                buf0[(r + 8 * mh) * LDSTR + nt * 16 + ncol] =
                    (_Float16)(acc[r] + bb);
        }
        waitDS();

        // ---- coalesced b128 store of the 16x128 f16 tile
#pragma unroll
        for (int rr = 0; rr < 8; ++rr) {
            int r = rr * 2 + mh;
            int e = e0 + r;
            if (e < E_) {
                v8h v = *(const v8h*)(buf0 + r * LDSTR + ncol * 8);
                *(v8h*)(Wout + (size_t)e * F_ + ncol * 8) = v;
            }
        }
    }
}

// ---------------------------------------------------------------------------
// out = x @ w (+ bias) (+ resid)   x:[N,128] f32, w:[128,128] f32, out f32
// ---------------------------------------------------------------------------
template <bool RESID>
__global__ void __launch_bounds__(256)
node_gemm_kernel(const float* __restrict__ x,
                 const float* __restrict__ w,
                 const float* __restrict__ bias,     // may be nullptr
                 const float* __restrict__ resid,    // used iff RESID
                 float* __restrict__ out,
                 int n_nodes) {
    extern __shared__ __align__(32) char smem_raw[];
    _Float16* wf = (_Float16*)smem_raw;           // 4*8*32*16 = 16384 halves
    const int tid = threadIdx.x, lane = tid & 31, warp = tid >> 5;
    _Float16* al = wf + 4 * 8 * 32 * 16 + warp * (16 * LDSTR);

    preload_b_frag(wf, w, 4, F_, tid, blockDim.x);
    __syncthreads();

    int wave   = (blockIdx.x * blockDim.x + tid) >> 5;
    int nwave  = (gridDim.x * blockDim.x) >> 5;
    int ntiles = (n_nodes + 15) >> 4;
    const int m = lane & 15, ch = lane >> 4;

    for (int t = wave; t < ntiles; t += nwave) {
        int n0 = t << 4;
        // stage 16x128 f32 -> f16 LDS (lane: row m, 64-col half ch)
        {
            int row = n0 + m; if (row >= n_nodes) row = n_nodes - 1;
            const float4* xr = (const float4*)(x + (size_t)row * F_) + ch * 16;
#pragma unroll
            for (int c2 = 0; c2 < 16; ++c2) {
                float4 v = xr[c2];
                v4h hv = {(_Float16)v.x, (_Float16)v.y, (_Float16)v.z, (_Float16)v.w};
                *(v4h*)(al + m * LDSTR + ch * 64 + c2 * 4) = hv;
            }
        }
        waitDS();
#pragma unroll
        for (int nt = 0; nt < 8; ++nt) {
            v8f acc = {0, 0, 0, 0, 0, 0, 0, 0};
#pragma unroll
            for (int kt = 0; kt < 4; ++kt) {
                v16h a = load_a_frag(al + kt * 32, LDSTR, lane);
                v16h b = load_b_frag(wf, kt, nt, lane);
                acc = wmma16(a, b, acc);
            }
            float bb = bias ? bias[nt * 16 + (lane & 15)] : 0.0f;
#pragma unroll
            for (int r = 0; r < 8; ++r) {
                int row = n0 + r + 8 * ch;
                if (row < n_nodes) {
                    size_t o = (size_t)row * F_ + nt * 16 + (lane & 15);
                    float val = acc[r] + bb;
                    if (RESID) val += resid[o];
                    out[o] = val;
                }
            }
        }
        waitDS();   // A reads done before next tile's staging overwrites al
    }
}

// ---------------------------------------------------------------------------
// agg[row[e]] += hw[col[e]] * W[e]   (f32 atomics, accumulator is L2-resident)
// ---------------------------------------------------------------------------
__global__ void message_kernel(const _Float16* __restrict__ Wf,
                               const float* __restrict__ hw,
                               const int* __restrict__ ei,
                               float* __restrict__ agg, int E_) {
    int idx = blockIdx.x * blockDim.x + threadIdx.x;
    if (idx >= E_ * 32) return;
    int e = idx >> 5, c = idx & 31;
    int r  = ei[e];
    int cl = ei[E_ + e];
    v4h   wv = *(const v4h*)(Wf + (size_t)e * F_ + c * 4);
    float4 hv = ((const float4*)(hw + (size_t)cl * F_))[c];
    float* dst = agg + (size_t)r * F_ + c * 4;
    unsafeAtomicAdd(dst + 0, hv.x * (float)wv[0]);
    unsafeAtomicAdd(dst + 1, hv.y * (float)wv[1]);
    unsafeAtomicAdd(dst + 2, hv.z * (float)wv[2]);
    unsafeAtomicAdd(dst + 3, hv.w * (float)wv[3]);
}

// ---------------------------------------------------------------------------
// graph readout: g = ssp(h@w_g1+b_g1)@w_g2 + b_g2 ; out[batch[n]] += g
// ---------------------------------------------------------------------------
__global__ void readout_kernel(const float* __restrict__ h,
                               const float* __restrict__ w_g1,  // [128,64]
                               const float* __restrict__ b_g1,  // [64]
                               const float* __restrict__ w_g2,  // [64]
                               const float* __restrict__ b_g2,  // [1]
                               const int* __restrict__ batch,
                               float* __restrict__ out, int n) {
    int wave  = (blockIdx.x * blockDim.x + threadIdx.x) >> 5;
    int lane  = threadIdx.x & 31;
    int nwave = (gridDim.x * blockDim.x) >> 5;
    for (int node = wave; node < n; node += nwave) {
        const float* hr = h + (size_t)node * H_;
        float s = 0.0f;
#pragma unroll
        for (int rep = 0; rep < 2; ++rep) {
            int j = lane + rep * 32;
            float dot = b_g1[j];
            for (int k = 0; k < H_; ++k) dot += hr[k] * w_g1[k * 64 + j];
            s += sspf(dot) * w_g2[j];
        }
        for (int off = 16; off > 0; off >>= 1) s += __shfl_down(s, off, 32);
        if (lane == 0) unsafeAtomicAdd(&out[batch[node]], s + b_g2[0]);
    }
}

// ---------------------------------------------------------------------------
extern "C" void kernel_launch(void* const* d_in, const int* in_sizes, int n_in,
                              void* d_out, int out_size, void* d_ws, size_t ws_size,
                              hipStream_t stream) {
    const int*   atom_q = (const int*)d_in[0];
    const float* coords = (const float*)d_in[1];
    const int*   ei     = (const int*)d_in[2];
    const int*   batch  = (const int*)d_in[3];
    const float* emb    = (const float*)d_in[4];
    const float* w_e1   = (const float*)d_in[5];
    const float* b_e1   = (const float*)d_in[6];
    const float* w_e2   = (const float*)d_in[7];
    const float* b_e2   = (const float*)d_in[8];
    const float* w_el   = (const float*)d_in[9];
    const float* w_nm   = (const float*)d_in[10];
    const float* b_nm   = (const float*)d_in[11];
    const float* w_g1   = (const float*)d_in[12];
    const float* b_g1   = (const float*)d_in[13];
    const float* w_g2   = (const float*)d_in[14];
    const float* b_g2   = (const float*)d_in[15];

    const int N = in_sizes[0];
    const int E = in_sizes[2] / 2;
    float* out = (float*)d_out;

    // workspace layout
    char* ws = (char*)d_ws;
    size_t off = 0;
    _Float16* Wf = (_Float16*)(ws + off);
    off += (size_t)E * F_ * sizeof(_Float16); off = (off + 255) & ~(size_t)255;
    float* h   = (float*)(ws + off);
    off += (size_t)N * H_ * sizeof(float);    off = (off + 255) & ~(size_t)255;
    float* hw  = (float*)(ws + off);
    off += (size_t)N * F_ * sizeof(float);    off = (off + 255) & ~(size_t)255;
    float* agg = (float*)(ws + off);

    const size_t FILTER_SMEM =
        (size_t)(2 * 8 * 32 * 16 + 4 * 8 * 32 * 16 + WARPS * 2 * 16 * LDSTR) *
        sizeof(_Float16);
    const size_t GEMM_SMEM =
        (size_t)(4 * 8 * 32 * 16 + WARPS * 16 * LDSTR) * sizeof(_Float16);

    // 1) node embeddings
    gather_emb_kernel<<<(N * 32 + 255) / 256, 256, 0, stream>>>(atom_q, emb, h, N);

    // 2) edge filters (WMMA, f16 output)
    filter_kernel<<<1024, 256, FILTER_SMEM, stream>>>(
        coords, ei, w_e1, b_e1, w_e2, b_e2, Wf, E);

    // 3) interactions
    for (int it = 0; it < 2; ++it) {
        node_gemm_kernel<false><<<256, 256, GEMM_SMEM, stream>>>(
            h, w_el, nullptr, nullptr, hw, N);
        zero_f32_kernel<<<(N * F_ + 255) / 256, 256, 0, stream>>>(agg, N * F_);
        message_kernel<<<(E * 32 + 255) / 256, 256, 0, stream>>>(Wf, hw, ei, agg, E);
        node_gemm_kernel<true><<<256, 256, GEMM_SMEM, stream>>>(
            agg, w_nm, b_nm, h, h, N);
    }

    // 4) readout
    zero_f32_kernel<<<(out_size + 255) / 256, 256, 0, stream>>>(out, out_size);
    readout_kernel<<<(N + 7) / 8, 256, 0, stream>>>(
        h, w_g1, b_g1, w_g2, b_g2, batch, out, N);
}
